// IndividualRNNAutoEncoder_70351564309222
// MI455X (gfx1250) — matrix operand order repaired and checked by reading, not compiled, not run on hardware
//
#include <hip/hip_runtime.h>
#include <cstdint>
#include <cstddef>

// ---------------- problem constants ----------------
#define Vdim 4
#define Hdim 512
#define Edim 1024
#define Bdim 64
#define Tdim 256
#define NBLK 32
#define TPB  256

static constexpr int HSZ = Bdim * Hdim;   // 32768 bf16 elems per hidden buffer
static constexpr int ASZ = Bdim * Edim;   // 65536 bf16 elems per accumulator buffer

typedef unsigned short ushort_t;
typedef __attribute__((ext_vector_type(16))) __bf16 v16bf;
typedef __attribute__((ext_vector_type(8)))  float  v8f;
typedef __attribute__((ext_vector_type(4)))  unsigned int v4u;
typedef __attribute__((ext_vector_type(4)))  int v4i;
typedef __attribute__((ext_vector_type(8)))  int v8i;

union V16U { v16bf v; uint4 q[2]; };

#if __has_builtin(__builtin_amdgcn_tensor_load_to_lds) && __has_builtin(__builtin_amdgcn_s_wait_tensorcnt)
#define USE_TDM 1
#if __has_include(<hip/amd_detail/amd_gfx1250_TDM.h>)
#define TDM_6ARG 1
#endif
#endif

// ---------------- device helpers ----------------
__device__ __forceinline__ ushort_t f2bf(float f) {
    unsigned int u = __float_as_uint(f);
    unsigned int r = u + 0x7FFFu + ((u >> 16) & 1u);   // RNE
    return (ushort_t)(r >> 16);
}

__device__ __forceinline__ v8f wmma_bf(v16bf a, v16bf b, v8f c) {
    return __builtin_amdgcn_wmma_f32_16x16x32_bf16(false, a, false, b, (short)0, c, false, false);
}

// load one packed fragment ([lane][16 contiguous bf16]) from global
__device__ __forceinline__ v16bf load_packed_g(const ushort_t* base, int lane) {
    V16U u;
    const uint4* p = (const uint4*)base + lane * 2;
    u.q[0] = p[0];
    u.q[1] = p[1];
    return u.v;
}

#ifdef USE_TDM
// Issue one TDM 2D tile load: nrows rows of row_u64s 8-byte units from global
// (row stride stride_u64s 8-byte units) into contiguous LDS at lds_dst.
// D# layout per CDNA5 ISA (groups 2/3 zero => 2D tensor).
__device__ __forceinline__ void tdm_load_2d(const void* gsrc, void* lds_dst,
                                            unsigned row_u64s, unsigned nrows,
                                            unsigned stride_u64s) {
    unsigned long long ga = (unsigned long long)(size_t)gsrc;
    unsigned lds_off = (unsigned)(size_t)lds_dst;   // LDS aperture: low 32 bits = LDS byte offset
    v4u g0;
    g0[0] = 1u;                                             // count=1 (valid user descriptor)
    g0[1] = lds_off;                                        // lds_addr (bytes)
    g0[2] = (unsigned)(ga & 0xFFFFFFFFu);                   // global_addr[31:0]
    g0[3] = (unsigned)((ga >> 32) & 0x01FFFFFFu) | (2u << 30);  // global_addr[56:32] | type=2
    v8i g1;
    g1[0] = (int)(3u << 16);                                // workgroup_mask=0, data_size=3 (8B)
    g1[1] = (int)((row_u64s & 0xFFFFu) << 16);              // tensor_dim0[15:0]
    g1[2] = (int)(((row_u64s >> 16) & 0xFFFFu) |            // tensor_dim0[31:16]
                  ((nrows & 0xFFFFu) << 16));               // tensor_dim1[15:0]
    g1[3] = (int)(((nrows >> 16) & 0xFFFFu) |               // tensor_dim1[31:16]
                  ((row_u64s & 0xFFFFu) << 16));            // tile_dim0
    g1[4] = (int)(nrows & 0xFFFFu);                         // tile_dim1 (tile_dim2=0)
    g1[5] = (int)stride_u64s;                               // tensor_dim0_stride[31:0]
    g1[6] = 0;                                              // stride0 hi | dim1_stride lo
    g1[7] = 0;
    v4i z4 = {0, 0, 0, 0};
#ifdef TDM_6ARG
    v8i z8 = {0, 0, 0, 0, 0, 0, 0, 0};
    __builtin_amdgcn_tensor_load_to_lds(g0, g1, z4, z4, z8, 0);
#else
    __builtin_amdgcn_tensor_load_to_lds(g0, g1, z4, z4, 0);
#endif
}
#endif

struct Ptrs {
    // packed bf16 weights (B-fragment layout) and packed x (A-fragment layout)
    const ushort_t *Wih0p;
    const ushort_t *encWihp0, *encWihp1;
    const ushort_t *encWhhp0, *encWhhp1, *encWhhp2;
    const ushort_t *Wyap, *Waap;
    const ushort_t *decWih0p;
    const ushort_t *decWihp0, *decWihp1;
    const ushort_t *decWhhp0, *decWhhp1, *decWhhp2;
    const ushort_t *Wprojp;
    const ushort_t *xpack;
    // combined f32 biases
    const float *benc;   // [3][Hdim]
    const float *bacc;   // [Edim]
    const float *bdec;   // [3][Hdim]
    // bf16 state (ping-pong canonical [64][K])
    ushort_t *henc0, *henc1, *henc2;   // each [2][HSZ]
    ushort_t *astate;                  // [2][ASZ]
    ushort_t *hdec0, *hdec1, *hdec2;   // each [2][HSZ]
    float    *encproj;                 // [64][Hdim] f32
    float    *dout;                    // [B][T][V] f32
    int      *bar;                     // [counter, generation]
};

// global sense barrier
__device__ __forceinline__ void grid_sync(int* bar) {
    __threadfence();
    __syncthreads();
    if (threadIdx.x == 0) {
        int gen = atomicAdd(&bar[1], 0);
        int arrived = atomicAdd(&bar[0], 1);
        if (arrived == NBLK - 1) {
            atomicExch(&bar[0], 0);
            __threadfence();
            atomicAdd(&bar[1], 1);
        } else {
            while (atomicAdd(&bar[1], 0) == gen) {
                __builtin_amdgcn_s_sleep(2);
            }
        }
    }
    __syncthreads();
    __threadfence();
}

// GEMM over Kt k-tiles: A from canonical bf16 [64][astride] global buffer,
// B weight fragments staged through LDS in chunks of <=8 k-tiles via TDM
// (fallback: batched register staging). Block owns `ntcount` consecutive
// n-tiles at ntbase; this wave consumes n-tile (ntbase+ntl), m-tile mt.
__device__ __forceinline__ void gemm_lds(v8f& c,
                                         const ushort_t* A, int astride,
                                         const ushort_t* Bp, int Kt,
                                         int mt, int ntbase, int ntl, int ntcount,
                                         int lane, int tid, ushort_t* smem, bool act) {
    const int gwl = tid >> 5;
    const ushort_t* hb = A + (size_t)(mt * 16 + (lane & 15)) * astride + ((lane >> 4) << 4);
    for (int k0 = 0; k0 < Kt; k0 += 8) {
        int nk = Kt - k0; if (nk > 8) nk = 8;
        __syncthreads();
#ifdef USE_TDM
        if (gwl == 0) {
            // one 2D DMA: ntcount rows of nk*512 bf16, row stride Kt*512 bf16
            tdm_load_2d(Bp + ((size_t)ntbase * Kt + k0) * 512, smem,
                        (unsigned)(nk * 128), (unsigned)ntcount, (unsigned)(Kt * 128));
            __builtin_amdgcn_s_wait_tensorcnt(0);
        }
#else
        {   // batched register staging (all loads issued before stores)
            const int nvec = ntcount * nk * 64;       // 512 bf16 = 64 uint4 per frag
            uint4* d = (uint4*)smem;
            const uint4* s4 = (const uint4*)Bp;
            uint4 tmp[4];
            int cnt = 0;
            for (int i = tid; i < nvec; i += TPB) {
                int seg = i / (nk * 64);
                int rem = i - seg * (nk * 64);
                tmp[cnt++] = s4[((size_t)(ntbase + seg) * Kt + k0) * 64 + rem];
            }
            cnt = 0;
            for (int i = tid; i < nvec; i += TPB) {
                int seg = i / (nk * 64);
                int rem = i - seg * (nk * 64);
                d[seg * 512 + rem] = tmp[cnt++];
            }
        }
#endif
        if (k0 + 8 < Kt) {   // prefetch next chunk into GL2 (global_prefetch)
            __builtin_prefetch((const void*)((const uint4*)Bp +
                               ((size_t)ntbase * Kt + k0 + 8) * 64 + tid), 0, 1);
        }
        __syncthreads();
        if (act) {
            for (int kk = 0; kk < nk; ++kk) {
                int kt = k0 + kk;
                V16U a;
                const uint4* ap = (const uint4*)(hb + kt * 32);
                a.q[0] = ap[0];          // K slots 0..7 of this lane-group
                a.q[1] = ap[1];          // K slots +8..+15
                V16U b;
                const uint4* bp4 = (const uint4*)(smem + (size_t)(ntl * 8 + kk) * 512) + lane * 2;
                b.q[0] = bp4[0];
                b.q[1] = bp4[1];
                c = wmma_bf(a.v, b.v, c);
            }
        }
    }
}

__device__ __forceinline__ v8f bias_init(const float* bias, int nt, int lane) {
    float bv = bias[nt * 16 + (lane & 15)];
    v8f c;
    for (int r = 0; r < 8; ++r) c[r] = bv;
    return c;
}

__device__ __forceinline__ v8f cinit_from(const float* src, int stride, int mt, int nt, int lane) {
    int n  = nt * 16 + (lane & 15);
    int mb = mt * 16 + ((lane >> 4) << 3);
    v8f c;
    for (int r = 0; r < 8; ++r) c[r] = src[(size_t)(mb + r) * stride + n];
    return c;
}

__device__ __forceinline__ void store_tanh(ushort_t* dst, int stride, int mt, int nt, int lane, v8f c) {
    int n  = nt * 16 + (lane & 15);
    int mb = mt * 16 + ((lane >> 4) << 3);
    for (int r = 0; r < 8; ++r) {
        dst[(size_t)(mb + r) * stride + n] = f2bf(tanhf(c[r]));
    }
}

__device__ __forceinline__ void store_f32(float* dst, int stride, int mt, int nt, int lane, v8f c) {
    int n  = nt * 16 + (lane & 15);
    int mb = mt * 16 + ((lane >> 4) << 3);
    for (int r = 0; r < 8; ++r) {
        dst[(size_t)(mb + r) * stride + n] = c[r];
    }
}

// ---------------- pack kernels ----------------
// B-fragment layout (16-bit 32x16 K-major): lane = ((k>>4)&1)*16 + n16, slot = k&15
__global__ void pack_w(const float* __restrict__ W, ushort_t* __restrict__ dst,
                       int N, int K, int Nt, int Kt) {
    int p = blockIdx.x * blockDim.x + threadIdx.x;
    int total = Nt * Kt * 512;
    if (p >= total) return;
    int frag = p >> 9, within = p & 511;
    int lane = within >> 4, slot = within & 15;
    int kt = frag % Kt, nt = frag / Kt;
    int k = kt * 32 + ((lane >> 4) << 4) + slot;
    int n = nt * 16 + (lane & 15);
    float v = (n < N && k < K) ? W[(size_t)n * K + k] : 0.0f;
    dst[p] = f2bf(v);
}

// A-fragment layout (16-bit 16x32): lane = ((k>>4)&1)*16 + m16, slot = k&15
// packs x[:,t,:] (K=4, padded to 32) as one A fragment per (t, m-tile)
__global__ void pack_x(const float* __restrict__ x, ushort_t* __restrict__ dst) {
    int p = blockIdx.x * blockDim.x + threadIdx.x;
    int total = Tdim * 4 * 512;
    if (p >= total) return;
    int frag = p >> 9, within = p & 511;
    int lane = within >> 4, slot = within & 15;
    int t = frag >> 2, mt = frag & 3;
    int k = ((lane >> 4) << 4) + slot;
    int m = mt * 16 + (lane & 15);
    float v = (k < Vdim) ? x[((size_t)m * Tdim + t) * Vdim + k] : 0.0f;
    dst[p] = f2bf(v);
}

__global__ void combine_bias(const float* ebih, const float* ebhh,
                             const float* bya, const float* baa,
                             const float* dbih, const float* dbhh,
                             float* benc, float* bacc, float* bdec) {
    int i = blockIdx.x * blockDim.x + threadIdx.x;
    if (i < 3 * Hdim) benc[i] = ebih[i] + ebhh[i];
    else if (i < 3 * Hdim + Edim) { int j = i - 3 * Hdim; bacc[j] = bya[j] + baa[j]; }
    else if (i < 6 * Hdim + Edim) { int j = i - (3 * Hdim + Edim); bdec[j] = dbih[j] + dbhh[j]; }
}

__global__ void init_state(ushort_t* h0, ushort_t* h1, ushort_t* h2,
                           ushort_t* d0, ushort_t* d1, ushort_t* d2,
                           ushort_t* a, int* bar) {
    int i = blockIdx.x * blockDim.x + threadIdx.x;
    if (i < HSZ) { h0[HSZ + i] = 0; h1[HSZ + i] = 0; h2[HSZ + i] = 0;
                   d0[HSZ + i] = 0; d1[HSZ + i] = 0; d2[HSZ + i] = 0; }
    if (i < ASZ) a[ASZ + i] = 0;
    if (i < 2)   bar[i] = 0;
}

// ---------------- persistent RNN kernel ----------------
__global__ __launch_bounds__(TPB) void rnn_persist(Ptrs P) {
    const int tid  = threadIdx.x;
    const int lane = tid & 31;
    const int gwl  = tid >> 5;            // wave in block 0..7
    const int ntbase = blockIdx.x * 2;
    const int mt  = gwl & 3;
    const int ntl = gwl >> 2;             // 0 or 1
    const int nt  = ntbase + ntl;
    const bool layer_blk = (blockIdx.x < 16);

    __shared__ ushort_t smem[2 * 8 * 512];  // 16KB B-fragment staging

    // ================= encoder =================
    #pragma unroll 1
    for (int t = 0; t < Tdim; ++t) {
        const int wb = t & 1, rb = wb ^ 1;
        // ---- layer 0: tanh(x_t Wih0^T + b + h0 Whh0^T) ----
        if (layer_blk) {
            v8f c = bias_init(P.benc + 0 * Hdim, nt, lane);
            {   // K=4 (padded to 32) x-part, single fragment, no staging needed
                v16bf a = load_packed_g(P.xpack + ((size_t)t * 4 + mt) * 512, lane);
                v16bf b = load_packed_g(P.Wih0p + (size_t)nt * 512, lane);
                c = wmma_bf(a, b, c);
            }
            gemm_lds(c, P.henc0 + rb * HSZ, Hdim, P.encWhhp0, 16, mt, ntbase, ntl, 2, lane, tid, smem, true);
            store_tanh(P.henc0 + wb * HSZ, Hdim, mt, nt, lane, c);
        }
        grid_sync(P.bar);
        // ---- layer 1 ----
        if (layer_blk) {
            v8f c = bias_init(P.benc + 1 * Hdim, nt, lane);
            gemm_lds(c, P.henc0 + wb * HSZ, Hdim, P.encWihp0, 16, mt, ntbase, ntl, 2, lane, tid, smem, true);
            gemm_lds(c, P.henc1 + rb * HSZ, Hdim, P.encWhhp1, 16, mt, ntbase, ntl, 2, lane, tid, smem, true);
            store_tanh(P.henc1 + wb * HSZ, Hdim, mt, nt, lane, c);
        }
        grid_sync(P.bar);
        // ---- layer 2 ----
        if (layer_blk) {
            v8f c = bias_init(P.benc + 2 * Hdim, nt, lane);
            gemm_lds(c, P.henc1 + wb * HSZ, Hdim, P.encWihp1, 16, mt, ntbase, ntl, 2, lane, tid, smem, true);
            gemm_lds(c, P.henc2 + rb * HSZ, Hdim, P.encWhhp2, 16, mt, ntbase, ntl, 2, lane, tid, smem, true);
            store_tanh(P.henc2 + wb * HSZ, Hdim, mt, nt, lane, c);
        }
        grid_sync(P.bar);
        // ---- accumulator: a = tanh(h2 Wya^T + b + a Waa^T), N=1024 (all 32 blocks) ----
        {
            v8f c = bias_init(P.bacc, nt, lane);
            gemm_lds(c, P.henc2 + wb * HSZ, Hdim, P.Wyap, 16, mt, ntbase, ntl, 2, lane, tid, smem, true);
            gemm_lds(c, P.astate + rb * ASZ, Edim, P.Waap, 32, mt, ntbase, ntl, 2, lane, tid, smem, true);
            store_tanh(P.astate + wb * ASZ, Edim, mt, nt, lane, c);
        }
        grid_sync(P.bar);
    }

    // ========== hoisted decoder input: enc_proj = encoding Wih0^T + bih0 + bhh0 ==========
    if (layer_blk) {
        v8f c = bias_init(P.bdec + 0 * Hdim, nt, lane);
        gemm_lds(c, P.astate + 1 * ASZ, Edim, P.decWih0p, 32, mt, ntbase, ntl, 2, lane, tid, smem, true);
        store_f32(P.encproj, Hdim, mt, nt, lane, c);
    }
    grid_sync(P.bar);

    // ================= decoder =================
    #pragma unroll 1
    for (int t = 0; t < Tdim; ++t) {
        const int wb = t & 1, rb = wb ^ 1;
        // ---- layer 0: tanh(enc_proj + h0 Whh0^T) (biases folded into enc_proj) ----
        if (layer_blk) {
            v8f c = cinit_from(P.encproj, Hdim, mt, nt, lane);
            gemm_lds(c, P.hdec0 + rb * HSZ, Hdim, P.decWhhp0, 16, mt, ntbase, ntl, 2, lane, tid, smem, true);
            store_tanh(P.hdec0 + wb * HSZ, Hdim, mt, nt, lane, c);
        }
        grid_sync(P.bar);
        // ---- layer 1 ----
        if (layer_blk) {
            v8f c = bias_init(P.bdec + 1 * Hdim, nt, lane);
            gemm_lds(c, P.hdec0 + wb * HSZ, Hdim, P.decWihp0, 16, mt, ntbase, ntl, 2, lane, tid, smem, true);
            gemm_lds(c, P.hdec1 + rb * HSZ, Hdim, P.decWhhp1, 16, mt, ntbase, ntl, 2, lane, tid, smem, true);
            store_tanh(P.hdec1 + wb * HSZ, Hdim, mt, nt, lane, c);
        }
        grid_sync(P.bar);
        // ---- layer 2 ----
        if (layer_blk) {
            v8f c = bias_init(P.bdec + 2 * Hdim, nt, lane);
            gemm_lds(c, P.hdec1 + wb * HSZ, Hdim, P.decWihp1, 16, mt, ntbase, ntl, 2, lane, tid, smem, true);
            gemm_lds(c, P.hdec2 + rb * HSZ, Hdim, P.decWhhp2, 16, mt, ntbase, ntl, 2, lane, tid, smem, true);
            store_tanh(P.hdec2 + wb * HSZ, Hdim, mt, nt, lane, c);
        }
        grid_sync(P.bar);
        // ---- projection + log_softmax (block 0 only; overlaps next step's L0 on other blocks;
        //      hdec2[wb] is not rewritten until t+2's L2, which is >=3 barriers away) ----
        if (blockIdx.x == 0) {
            bool act = (gwl < 4);
            int pmt = gwl & 3;
            v8f c; for (int r = 0; r < 8; ++r) c[r] = 0.0f;
            gemm_lds(c, P.hdec2 + wb * HSZ, Hdim, P.Wprojp, 16, pmt, 0, 0, 1, lane, tid, smem, act);
            if (act) {
                int n  = lane & 15;
                int mb = pmt * 16 + ((lane >> 4) << 3);
                for (int r = 0; r < 8; ++r) {
                    float v  = c[r];
                    float m1 = fmaxf(v, __shfl_xor(v, 1));
                    float mx = fmaxf(m1, __shfl_xor(m1, 2));
                    float e  = __expf(v - mx);
                    float s1 = e + __shfl_xor(e, 1);
                    float s  = s1 + __shfl_xor(s1, 2);
                    float o  = v - (mx + __logf(s));
                    if (n < Vdim) {
                        int m = mb + r;
                        P.dout[((size_t)m * Tdim + t) * Vdim + n] = o;
                    }
                }
            }
        }
    }
}

// ---------------- host launcher ----------------
extern "C" void kernel_launch(void* const* d_in, const int* in_sizes, int n_in,
                              void* d_out, int out_size, void* d_ws, size_t ws_size,
                              hipStream_t stream) {
    (void)in_sizes; (void)n_in; (void)out_size; (void)ws_size;

    const float* x     = (const float*)d_in[0];
    const float* eWih0 = (const float*)d_in[1];
    const float* eWih  = (const float*)d_in[2];
    const float* eWhh  = (const float*)d_in[3];
    const float* ebih  = (const float*)d_in[4];
    const float* ebhh  = (const float*)d_in[5];
    const float* Wya   = (const float*)d_in[6];
    const float* bya   = (const float*)d_in[7];
    const float* Waa   = (const float*)d_in[8];
    const float* baa   = (const float*)d_in[9];
    const float* dWih0 = (const float*)d_in[10];
    const float* dWih  = (const float*)d_in[11];
    const float* dWhh  = (const float*)d_in[12];
    const float* dbih  = (const float*)d_in[13];
    const float* dbhh  = (const float*)d_in[14];
    const float* Wproj = (const float*)d_in[15];

    char* base = (char*)d_ws;
    size_t off = 0;
    auto take = [&](size_t bytes) -> void* {
        void* p = base + off;
        off = (off + bytes + 255) & ~(size_t)255;
        return p;
    };

    Ptrs P;
    ushort_t* wih0p   = (ushort_t*)take(32 * 1 * 512 * 2);
    ushort_t* ewihp0  = (ushort_t*)take(32 * 16 * 512 * 2);
    ushort_t* ewihp1  = (ushort_t*)take(32 * 16 * 512 * 2);
    ushort_t* ewhhp0  = (ushort_t*)take(32 * 16 * 512 * 2);
    ushort_t* ewhhp1  = (ushort_t*)take(32 * 16 * 512 * 2);
    ushort_t* ewhhp2  = (ushort_t*)take(32 * 16 * 512 * 2);
    ushort_t* wyap    = (ushort_t*)take(64 * 16 * 512 * 2);
    ushort_t* waap    = (ushort_t*)take(64 * 32 * 512 * 2);
    ushort_t* dwih0p  = (ushort_t*)take(32 * 32 * 512 * 2);
    ushort_t* dwihp0  = (ushort_t*)take(32 * 16 * 512 * 2);
    ushort_t* dwihp1  = (ushort_t*)take(32 * 16 * 512 * 2);
    ushort_t* dwhhp0  = (ushort_t*)take(32 * 16 * 512 * 2);
    ushort_t* dwhhp1  = (ushort_t*)take(32 * 16 * 512 * 2);
    ushort_t* dwhhp2  = (ushort_t*)take(32 * 16 * 512 * 2);
    ushort_t* wprojp  = (ushort_t*)take(1 * 16 * 512 * 2);
    ushort_t* xpack   = (ushort_t*)take((size_t)Tdim * 4 * 512 * 2);
    float* benc    = (float*)take(3 * Hdim * 4);
    float* bacc    = (float*)take(Edim * 4);
    float* bdec    = (float*)take(3 * Hdim * 4);
    float* encproj = (float*)take((size_t)Bdim * Hdim * 4);
    ushort_t* henc0 = (ushort_t*)take(2 * HSZ * 2);
    ushort_t* henc1 = (ushort_t*)take(2 * HSZ * 2);
    ushort_t* henc2 = (ushort_t*)take(2 * HSZ * 2);
    ushort_t* astate= (ushort_t*)take(2 * ASZ * 2);
    ushort_t* hdec0 = (ushort_t*)take(2 * HSZ * 2);
    ushort_t* hdec1 = (ushort_t*)take(2 * HSZ * 2);
    ushort_t* hdec2 = (ushort_t*)take(2 * HSZ * 2);
    int* bar        = (int*)take(256);

    P.Wih0p = wih0p; P.encWihp0 = ewihp0; P.encWihp1 = ewihp1;
    P.encWhhp0 = ewhhp0; P.encWhhp1 = ewhhp1; P.encWhhp2 = ewhhp2;
    P.Wyap = wyap; P.Waap = waap; P.decWih0p = dwih0p;
    P.decWihp0 = dwihp0; P.decWihp1 = dwihp1;
    P.decWhhp0 = dwhhp0; P.decWhhp1 = dwhhp1; P.decWhhp2 = dwhhp2;
    P.Wprojp = wprojp; P.xpack = xpack;
    P.benc = benc; P.bacc = bacc; P.bdec = bdec;
    P.henc0 = henc0; P.henc1 = henc1; P.henc2 = henc2;
    P.astate = astate; P.hdec0 = hdec0; P.hdec1 = hdec1; P.hdec2 = hdec2;
    P.encproj = encproj; P.dout = (float*)d_out; P.bar = bar;

    auto gsz = [](int total) { return (total + TPB - 1) / TPB; };

    pack_w<<<gsz(32 * 1 * 512), TPB, 0, stream>>>(eWih0, wih0p, 512, 4, 32, 1);
    pack_w<<<gsz(32 * 16 * 512), TPB, 0, stream>>>(eWih + 0 * 512 * 512, ewihp0, 512, 512, 32, 16);
    pack_w<<<gsz(32 * 16 * 512), TPB, 0, stream>>>(eWih + 1 * 512 * 512, ewihp1, 512, 512, 32, 16);
    pack_w<<<gsz(32 * 16 * 512), TPB, 0, stream>>>(eWhh + 0 * 512 * 512, ewhhp0, 512, 512, 32, 16);
    pack_w<<<gsz(32 * 16 * 512), TPB, 0, stream>>>(eWhh + 1 * 512 * 512, ewhhp1, 512, 512, 32, 16);
    pack_w<<<gsz(32 * 16 * 512), TPB, 0, stream>>>(eWhh + 2 * 512 * 512, ewhhp2, 512, 512, 32, 16);
    pack_w<<<gsz(64 * 16 * 512), TPB, 0, stream>>>(Wya, wyap, 1024, 512, 64, 16);
    pack_w<<<gsz(64 * 32 * 512), TPB, 0, stream>>>(Waa, waap, 1024, 1024, 64, 32);
    pack_w<<<gsz(32 * 32 * 512), TPB, 0, stream>>>(dWih0, dwih0p, 512, 1024, 32, 32);
    pack_w<<<gsz(32 * 16 * 512), TPB, 0, stream>>>(dWih + 0 * 512 * 512, dwihp0, 512, 512, 32, 16);
    pack_w<<<gsz(32 * 16 * 512), TPB, 0, stream>>>(dWih + 1 * 512 * 512, dwihp1, 512, 512, 32, 16);
    pack_w<<<gsz(32 * 16 * 512), TPB, 0, stream>>>(dWhh + 0 * 512 * 512, dwhhp0, 512, 512, 32, 16);
    pack_w<<<gsz(32 * 16 * 512), TPB, 0, stream>>>(dWhh + 1 * 512 * 512, dwhhp1, 512, 512, 32, 16);
    pack_w<<<gsz(32 * 16 * 512), TPB, 0, stream>>>(dWhh + 2 * 512 * 512, dwhhp2, 512, 512, 32, 16);
    pack_w<<<gsz(1 * 16 * 512), TPB, 0, stream>>>(Wproj, wprojp, 4, 512, 1, 16);
    pack_x<<<gsz(Tdim * 4 * 512), TPB, 0, stream>>>(x, xpack);
    combine_bias<<<gsz(6 * Hdim + Edim), TPB, 0, stream>>>(ebih, ebhh, bya, baa, dbih, dbhh, benc, bacc, bdec);
    init_state<<<gsz(ASZ), TPB, 0, stream>>>(henc0, henc1, henc2, hdec0, hdec1, hdec2, astate, bar);

    rnn_persist<<<NBLK, TPB, 0, stream>>>(P);
}
